// MyTransformerEncoderLayer_60120952209824
// MI455X (gfx1250) — compile-verified, matrix operand order checked by reading, not run on hardware
//
#include <hip/hip_runtime.h>

// ---------------------------------------------------------------------------
// MI455X / gfx1250 encoder layer: bf16 WMMA + double-buffered TDM staging.
// B=128 S=77 D=1024 H=16 DH=64 F=4096 E=8 topK=2,  M = B*S = 9856 = 77*128
// ~580 GFLOP/call vs ~0.6 GB HBM -> compute bound; bf16 expert bank (134 MB)
// is L2-resident (192 MB L2); MoE kernel: one WG per batch element (M=80) so
// L2 weight traffic is 128 x 48 MB ~= 6 GB/call; TDM prefetches the next
// weight tile while WMMA consumes the current one (TENSORcnt in-order).
// ---------------------------------------------------------------------------

typedef __attribute__((ext_vector_type(16))) __bf16 v16bf;
typedef __attribute__((ext_vector_type(8)))  float  v8f;
typedef __attribute__((ext_vector_type(4)))  unsigned v4u;
typedef __attribute__((ext_vector_type(8)))  int      v8i;
typedef __attribute__((ext_vector_type(4)))  int      v4i;

#define DEV __device__ __forceinline__

DEV unsigned short f2bf(float f) {                 // round-to-nearest-even f32->bf16
  unsigned u = __builtin_bit_cast(unsigned, f);
  unsigned r = u + 0x7FFFu + ((u >> 16) & 1u);
  return (unsigned short)(r >> 16);
}
DEV float bf2f(unsigned short h) {
  unsigned u = ((unsigned)h) << 16;
  return __builtin_bit_cast(float, u);
}
DEV __bf16 bfbits(unsigned short h) { return __builtin_bit_cast(__bf16, h); }

// ---- WMMA fragment index maps (CDNA5 ISA 7.12.2, 16-bit A 16x32 / C 16x16) ----
DEV int kbase(int lane, int p) {
  return (lane < 16) ? ((p < 4) ? 0 : 8) : ((p < 4) ? 8 : 16);
}
DEV int crow(int lane, int r) { return r + ((lane < 16) ? 0 : 8); }
DEV int ccol(int lane)        { return lane & 15; }

// A fragment: 16x32 bf16 tile, row-major LDS buffer with leading dim LD (even).
template <int LD>
DEV v16bf load_a_frag(const unsigned short* buf, int row0, int col0) {
  const int lane = threadIdx.x & 31;
  const unsigned short* r = buf + (row0 + (lane & 15)) * LD + col0;
  v16bf a;
#pragma unroll
  for (int p = 0; p < 8; ++p) {
    int k = kbase(lane, p) + 2 * p;
    unsigned v = *(const unsigned*)(r + k);
    a[2 * p]     = bfbits((unsigned short)(v & 0xffffu));
    a[2 * p + 1] = bfbits((unsigned short)(v >> 16));
  }
  return a;
}

// B fragment: 32x16 bf16 tile B[k][n], row-major (k-major) LDS buffer.
template <int LD>
DEV v16bf load_b_frag(const unsigned short* buf, int k0, int n0) {
  const int lane = threadIdx.x & 31;
  const unsigned short* c = buf + k0 * LD + n0 + (lane & 15);
  v16bf b;
#pragma unroll
  for (int j = 0; j < 16; ++j) {
    int p = j >> 1, w = j & 1;
    int k = kbase(lane, p) + 2 * p + w;
    b[j] = bfbits(c[k * LD]);
  }
  return b;
}

DEV v8f wmma_bf16(v16bf a, v16bf b, v8f c) {
  return __builtin_amdgcn_wmma_f32_16x16x32_bf16(false, a, false, b, (short)0, c,
                                                 false, false);
}

// ---------------------------------------------------------------------------
// Tensor Data Mover: 2D bf16 tile (tile_w x tile_h elements) from a row-major
// tensor (row stride `stride` elements) into LDS.  Optional per-row padding:
// interval = 2^(pad_int+1) dwords, amount = pad_amt+1 dwords.  D# per CDNA5
// ISA ch.8; rows beyond tensor_h read as zero (free S=77->80 zero-fill).
// ---------------------------------------------------------------------------
DEV void tdm_load_2d(unsigned lds_off, const void* gaddr,
                     unsigned tensor_w, unsigned tensor_h, unsigned stride,
                     unsigned tile_w, unsigned tile_h,
                     int pad_en, unsigned pad_int, unsigned pad_amt) {
  unsigned long long ga = (unsigned long long)gaddr;
  v4u g0;
  g0[0] = 1u;                                          // count=1 (valid), no gather
  g0[1] = lds_off;                                     // lds_addr (bytes)
  g0[2] = (unsigned)ga;                                // global_addr[31:0]
  g0[3] = (unsigned)((ga >> 32) & 0x01ffffffu) | (2u << 30);  // addr[56:32]|type=2
  v8i g1;
  unsigned d0 = 1u << 16;                              // data_size=1 -> 2 bytes
  if (pad_en) d0 |= (1u << 20) | (pad_int << 22) | (pad_amt << 25);
  g1[0] = (int)d0;
  g1[1] = (int)((tensor_w & 0xffffu) << 16);           // tensor_dim0[15:0]
  g1[2] = (int)((tensor_w >> 16) | ((tensor_h & 0xffffu) << 16));
  g1[3] = (int)((tensor_h >> 16) | (tile_w << 16));    // tensor_dim1 hi | tile_dim0
  g1[4] = (int)tile_h;                                 // tile_dim1 (tile_dim2=0)
  g1[5] = (int)stride;                                 // tensor_dim0_stride[31:0]
  g1[6] = 0;
  g1[7] = 0;
  v4i z4 = {0, 0, 0, 0};
  v8i z8 = {0, 0, 0, 0, 0, 0, 0, 0};
  __builtin_amdgcn_tensor_load_to_lds(g0, g1, z4, z4, z8, 0);
}

// ---------------------------------------------------------------------------
// f32 -> bf16 weight conversion (optionally transposed: [R,C] -> [C,R])
// ---------------------------------------------------------------------------
__global__ __launch_bounds__(256) void k_cvt(const float* __restrict__ src,
                                             unsigned short* __restrict__ dst,
                                             int rows, int cols, int transpose) {
  long long n = (long long)rows * cols;
  for (long long i = blockIdx.x * (long long)blockDim.x + threadIdx.x; i < n;
       i += (long long)gridDim.x * blockDim.x) {
    long long r = i / cols, c = i - r * cols;
    unsigned short v = f2bf(src[i]);
    if (transpose) dst[c * (long long)rows + r] = v;
    else           dst[i] = v;
  }
}

// ---------------------------------------------------------------------------
// LayerNorm (one 8-wave block per row of length D=1024), bf16 output
// ---------------------------------------------------------------------------
__global__ __launch_bounds__(256) void k_ln(const float* __restrict__ x,
                                            const float* __restrict__ g,
                                            const float* __restrict__ b,
                                            unsigned short* __restrict__ o, int D) {
  const int row = blockIdx.x;
  const float* xr = x + (long long)row * D;
  __shared__ float red[2][8];
  const int tid = threadIdx.x, lane = tid & 31, wv = tid >> 5;
  float s = 0.f, s2 = 0.f;
  for (int i = tid; i < D; i += 256) { float v = xr[i]; s += v; s2 += v * v; }
  for (int off = 16; off; off >>= 1) { s += __shfl_xor(s, off, 32); s2 += __shfl_xor(s2, off, 32); }
  if (lane == 0) { red[0][wv] = s; red[1][wv] = s2; }
  __syncthreads();
  if (tid == 0) {
    float ts = 0.f, ts2 = 0.f;
    for (int i = 0; i < 8; ++i) { ts += red[0][i]; ts2 += red[1][i]; }
    float mean = ts / D;
    red[0][0] = mean;
    red[1][0] = ts2 / D - mean * mean;
  }
  __syncthreads();
  const float mean = red[0][0];
  const float rstd = rsqrtf(red[1][0] + 1e-5f);
  unsigned short* orow = o + (long long)row * D;
  for (int i = tid; i < D; i += 256)
    orow[i] = f2bf((xr[i] - mean) * rstd * g[i] + b[i]);
}

// ---------------------------------------------------------------------------
// Tiled bf16 WMMA GEMM: C[M,N] = A[M,K] * B[K,N] (+bias, +residual)
// BM=128 BN=128 BK=64, 8 waves in 2x4, 16 WMMA per wave per stage.
// ---------------------------------------------------------------------------
template <bool RESID_, bool OUTBF_>
__global__ __launch_bounds__(256) void k_gemm(const unsigned short* __restrict__ A,
                                              const unsigned short* __restrict__ Bm,
                                              const float* __restrict__ bias,
                                              const float* __restrict__ resid,
                                              float* __restrict__ Cf,
                                              unsigned short* __restrict__ Cb,
                                              int M, int N, int K) {
  __shared__ __align__(16) unsigned short sA[128 * 64];   // 16 KB
  __shared__ __align__(16) unsigned short sB[64 * 136];   // 17.4 KB (pad)
  const int tid = threadIdx.x, lane = tid & 31, wv = tid >> 5;
  const int wm = wv >> 2, wn = wv & 3;
  const int bm = blockIdx.y * 128, bn = blockIdx.x * 128;
  (void)M;
  v8f acc[4][2] = {};
  for (int k0 = 0; k0 < K; k0 += 64) {
    { // stage A: thread = (row, 32-col half)
      int row = tid >> 1, half = tid & 1;
      const uint4* s4 = (const uint4*)(A + (long long)(bm + row) * K + k0 + half * 32);
      uint4* d = (uint4*)(sA + row * 64 + half * 32);
      d[0] = s4[0]; d[1] = s4[1]; d[2] = s4[2]; d[3] = s4[3];
    }
    { // stage B: thread = (row, 32-col quarter)
      int row = tid >> 2, seg = tid & 3;
      const uint4* s4 = (const uint4*)(Bm + (long long)(k0 + row) * N + bn + seg * 32);
      uint4* d = (uint4*)(sB + row * 136 + seg * 32);
      d[0] = s4[0]; d[1] = s4[1]; d[2] = s4[2]; d[3] = s4[3];
    }
    if (k0 + 64 < K) {  // gfx1250 global_prefetch_b8 for next k-stage
      __builtin_prefetch(A + (long long)(bm + (tid >> 1)) * K + k0 + 64, 0, 1);
      __builtin_prefetch(Bm + (long long)(k0 + 64 + (tid >> 2)) * N + bn, 0, 1);
    }
    __syncthreads();
#pragma unroll
    for (int kk = 0; kk < 64; kk += 32) {
      v16bf af[4], bf[2];
#pragma unroll
      for (int i = 0; i < 4; ++i) af[i] = load_a_frag<64>(sA, wm * 64 + i * 16, kk);
#pragma unroll
      for (int j = 0; j < 2; ++j) bf[j] = load_b_frag<136>(sB, kk, wn * 32 + j * 16);
#pragma unroll
      for (int i = 0; i < 4; ++i)
#pragma unroll
        for (int j = 0; j < 2; ++j) acc[i][j] = wmma_bf16(af[i], bf[j], acc[i][j]);
    }
    __syncthreads();
  }
#pragma unroll
  for (int i = 0; i < 4; ++i)
#pragma unroll
    for (int j = 0; j < 2; ++j) {
      int col = bn + wn * 32 + j * 16 + ccol(lane);
#pragma unroll
      for (int r = 0; r < 8; ++r) {
        int row = bm + wm * 64 + i * 16 + crow(lane, r);
        float v = acc[i][j][r] + bias[col];
        if (RESID_) v += resid[(long long)row * N + col];
        if (OUTBF_) Cb[(long long)row * N + col] = f2bf(v);
        else        Cf[(long long)row * N + col] = v;
      }
    }
}

// ---------------------------------------------------------------------------
// Attention: one block per (b,h). S=77 padded to 80 (M/N) and 96 (K of P@V).
// ---------------------------------------------------------------------------
__global__ __launch_bounds__(256) void k_attn(const unsigned short* __restrict__ qkv,
                                              unsigned short* __restrict__ ao) {
  const int b = blockIdx.x >> 4;
  const int h = blockIdx.x & 15;
  __shared__ __align__(16) unsigned short sQ[80 * 68];
  __shared__ __align__(16) unsigned short sK[80 * 68];
  __shared__ __align__(16) unsigned short sV[96 * 68];
  __shared__ __align__(16) unsigned short sP[80 * 100];
  const int tid = threadIdx.x, lane = tid & 31, wv = tid >> 5;
  const long long base = ((long long)b * 77) * 3072 + h * 64;

  for (int i = tid; i < 80 * 32; i += 256) {
    int row = i >> 5, c2 = (i & 31) * 2;
    unsigned q = 0u, k = 0u;
    if (row < 77) {
      const unsigned short* p = qkv + base + (long long)row * 3072;
      q = *(const unsigned*)(p + c2);
      k = *(const unsigned*)(p + 1024 + c2);
    }
    *(unsigned*)(sQ + row * 68 + c2) = q;
    *(unsigned*)(sK + row * 68 + c2) = k;
  }
  for (int i = tid; i < 96 * 32; i += 256) {
    int row = i >> 5, c2 = (i & 31) * 2;
    unsigned v = 0u;
    if (row < 77) v = *(const unsigned*)(qkv + base + (long long)row * 3072 + 2048 + c2);
    *(unsigned*)(sV + row * 68 + c2) = v;
  }
  __syncthreads();

  if (wv < 5) {                                          // scores + softmax
    v16bf a0 = load_a_frag<68>(sQ, wv * 16, 0);
    v16bf a1 = load_a_frag<68>(sQ, wv * 16, 32);
    v8f sc[5] = {};
#pragma unroll
    for (int tn = 0; tn < 5; ++tn) {
      sc[tn] = wmma_bf16(a0, load_a_frag<68>(sK, tn * 16, 0),  sc[tn]);
      sc[tn] = wmma_bf16(a1, load_a_frag<68>(sK, tn * 16, 32), sc[tn]);
    }
    const int cb = ccol(lane);
#pragma unroll
    for (int r = 0; r < 8; ++r) {
      float m = -3.0e38f;
#pragma unroll
      for (int tn = 0; tn < 5; ++tn)
        if (tn * 16 + cb < 77) m = fmaxf(m, sc[tn][r] * 0.125f);
      for (int off = 8; off; off >>= 1) m = fmaxf(m, __shfl_xor(m, off, 32));
      float ex[5], sum = 0.f;
#pragma unroll
      for (int tn = 0; tn < 5; ++tn) {
        ex[tn] = (tn * 16 + cb < 77) ? __expf(sc[tn][r] * 0.125f - m) : 0.f;
        sum += ex[tn];
      }
      for (int off = 8; off; off >>= 1) sum += __shfl_xor(sum, off, 32);
      int row = wv * 16 + crow(lane, r);
      float inv = (row < 77) ? (1.f / sum) : 0.f;
#pragma unroll
      for (int tn = 0; tn < 5; ++tn)
        sP[row * 100 + tn * 16 + cb] = f2bf(ex[tn] * inv);
      sP[row * 100 + 80 + cb] = 0;                       // zero K-pad cols 80..95
    }
  }
  __syncthreads();

  for (int t = wv; t < 20; t += 8) {                     // out = P @ V
    int tm = t >> 2, tn = t & 3;
    v8f acc = {};
#pragma unroll
    for (int kk = 0; kk < 3; ++kk) {
      v16bf a = load_a_frag<100>(sP, tm * 16, kk * 32);
      v16bf bb = load_b_frag<68>(sV, kk * 32, tn * 16);
      acc = wmma_bf16(a, bb, acc);
    }
#pragma unroll
    for (int r = 0; r < 8; ++r) {
      int row = tm * 16 + crow(lane, r);
      if (row < 77)
        ao[((long long)b * 77 + row) * 1024 + h * 64 + tn * 16 + ccol(lane)] =
            f2bf(acc[r]);
    }
  }
}

// ---------------------------------------------------------------------------
// Router: wave e computes logit[b,e]; thread 0 does top-2 + softmax gates.
// ---------------------------------------------------------------------------
__global__ __launch_bounds__(256) void k_router(const unsigned short* __restrict__ ffn,
                                                const float* __restrict__ rw,
                                                int* __restrict__ eidx,
                                                float* __restrict__ gates) {
  const int b = blockIdx.x;
  __shared__ float sl[8];
  const int tid = threadIdx.x, lane = tid & 31, wv = tid >> 5;
  const unsigned short* x = ffn + (long long)b * 77 * 1024;  // token 0
  const float* w = rw + wv * 1024;
  float s = 0.f;
  for (int i = lane; i < 1024; i += 32) s += bf2f(x[i]) * w[i];
  for (int off = 16; off; off >>= 1) s += __shfl_xor(s, off, 32);
  if (lane == 0) sl[wv] = s;
  __syncthreads();
  if (tid == 0) {
    int i0 = 0; float v0 = sl[0];
    for (int i = 1; i < 8; ++i) if (sl[i] > v0) { v0 = sl[i]; i0 = i; }
    int i1 = (i0 == 0) ? 1 : 0; float v1 = sl[i1];
    for (int i = 0; i < 8; ++i) if (i != i0 && sl[i] > v1) { v1 = sl[i]; i1 = i; }
    float e1 = __expf(v1 - v0), inv = 1.f / (1.f + e1);
    eidx[b * 2] = i0;  eidx[b * 2 + 1] = i1;
    gates[b * 2] = inv; gates[b * 2 + 1] = e1 * inv;
  }
}

// ---------------------------------------------------------------------------
// Fused MoE + shared-expert FFN.  One workgroup per batch element (M=80,
// 5 m-tiles), so each expert weight matrix streams from L2 once per b.
// Double-buffered TDM pipeline: wave 0 issues stage k+1's tensor_load_to_lds
// into the alternate LDS buffer, then s_wait_tensorcnt <= (#in-flight next)
// so stage k is complete while k+1 streams behind the WMMAs (TENSORcnt
// completes in order).  LDS: X 2x5 KB + W 2x16.6 KB + h 20.5 KB = 64 000 B.
// ---------------------------------------------------------------------------
__global__ __launch_bounds__(256) void k_moe(const unsigned short* __restrict__ ffn,
                                             const unsigned short* __restrict__ ew1b,
                                             const float* __restrict__ eb1,
                                             const unsigned short* __restrict__ ew2b,
                                             const float* __restrict__ eb2,
                                             const unsigned short* __restrict__ sw1b,
                                             const float* __restrict__ sb1,
                                             const unsigned short* __restrict__ sw2b,
                                             const float* __restrict__ sb2,
                                             const int* __restrict__ eidx,
                                             const float* __restrict__ gates,
                                             float* __restrict__ out) {
  const int b = blockIdx.x;
  __shared__ __align__(16) unsigned short sXk[2][80 * 32];  // X k-tiles (LD 32)
  __shared__ __align__(16) unsigned short sW[2][32 * 260];  // W tiles (LD 132/260)
  __shared__ __align__(16) unsigned short sH[80 * 128];     // h chunk  (LD 128)
  const int tid = threadIdx.x, lane = tid & 31, wv = tid >> 5;
  const int cb = ccol(lane);
  const unsigned offX0 = (unsigned)(unsigned long long)(void*)sXk[0];
  const unsigned offX1 = (unsigned)(unsigned long long)(void*)sXk[1];
  const unsigned offW0 = (unsigned)(unsigned long long)(void*)sW[0];
  const unsigned offW1 = (unsigned)(unsigned long long)(void*)sW[1];
  const unsigned short* xbase = ffn + (long long)b * 77 * 1024;

  v8f eo[5][8] = {};   // [m-tile][n-tile]; n-tile t: col (t>>1)*256 + wv*32 + (t&1)*16
  for (int xp = 0; xp < 3; ++xp) {
    const unsigned short *w1, *w2; const float *b1, *b2; float g;
    if (xp < 2) {
      int e = eidx[b * 2 + xp]; g = gates[b * 2 + xp];
      w1 = ew1b + (long long)e * 1024 * 4096; b1 = eb1 + e * 4096;
      w2 = ew2b + (long long)e * 4096 * 1024; b2 = eb2 + e * 1024;
    } else { g = 1.f; w1 = sw1b; b1 = sb1; w2 = sw2b; b2 = sb2; }

    for (int fc = 0; fc < 4096; fc += 128) {
      // ---- GEMM1: h[80,128] = x[80,1024] @ W1[1024, fc:fc+128] ----
      v8f hacc[5] = {};
      for (int ks = 0; ks < 32; ++ks) {                 // k0 = ks*32
        int pb = ks & 1;
        __syncthreads();                 // prior consumers of target buffers done
        if (wv == 0) {                   // double-buffered TDM pipeline
          if (ks == 0) {                 // prologue: stage 0 into buffer 0
            tdm_load_2d(offX0, xbase, 1024, 77, 1024, 32, 80, 0, 0, 0);
            tdm_load_2d(offW0, w1 + fc, 4096, 1024, 4096, 128, 32, 1, 5, 1);
          }
          if (ks + 1 < 32) {             // prefetch stage ks+1 into alt buffer
            int kn = (ks + 1) * 32;
            tdm_load_2d(pb ? offX0 : offX1, xbase + kn, 1024, 77, 1024,
                        32, 80, 0, 0, 0);
            tdm_load_2d(pb ? offW0 : offW1, w1 + (long long)kn * 4096 + fc,
                        4096, 1024, 4096, 128, 32, 1, 5, 1);
            __builtin_amdgcn_s_wait_tensorcnt(2);  // stage ks done, ks+1 in flight
          } else {
            __builtin_amdgcn_s_wait_tensorcnt(0);
          }
        }
        __syncthreads();                 // stage ks data visible to all waves
        v16bf bb = load_b_frag<132>(sW[pb], 0, wv * 16);
#pragma unroll
        for (int m = 0; m < 5; ++m)
          hacc[m] = wmma_bf16(load_a_frag<32>(sXk[pb], m * 16, 0), bb, hacc[m]);
      }
      __syncthreads();
      // h epilogue: +bias, relu, pre-scale by gate (g*(h@W2) == (g*h)@W2)
#pragma unroll
      for (int m = 0; m < 5; ++m)
#pragma unroll
        for (int r = 0; r < 8; ++r) {
          float v = hacc[m][r] + b1[fc + wv * 16 + cb];
          v = v > 0.f ? v : 0.f;
          sH[(m * 16 + crow(lane, r)) * 128 + wv * 16 + cb] = f2bf(v * g);
        }
      __syncthreads();
      // ---- GEMM2: eo += h[80,128] @ W2[fc:fc+128, 0:1024] ----
      for (int st = 0; st < 16; ++st) {                 // j = st>>2, s4 = st&3
        int pb = st & 1;
        __syncthreads();
        if (wv == 0) {
          if (st == 0)
            tdm_load_2d(offW0, w2 + (long long)fc * 1024, 1024, 4096, 1024,
                        256, 32, 1, 6, 1);
          if (st + 1 < 16) {
            int jn = (st + 1) >> 2, sn = (st + 1) & 3;
            tdm_load_2d(pb ? offW0 : offW1,
                        w2 + (long long)(fc + jn * 32) * 1024 + sn * 256,
                        1024, 4096, 1024, 256, 32, 1, 6, 1);
            __builtin_amdgcn_s_wait_tensorcnt(1);
          } else {
            __builtin_amdgcn_s_wait_tensorcnt(0);
          }
        }
        __syncthreads();
        int j = st >> 2, s4 = st & 3;
#pragma unroll
        for (int m = 0; m < 5; ++m) {
          v16bf a2 = load_a_frag<128>(sH, m * 16, j * 32);
          eo[m][s4 * 2]     = wmma_bf16(a2, load_b_frag<260>(sW[pb], 0, wv * 32),
                                        eo[m][s4 * 2]);
          eo[m][s4 * 2 + 1] = wmma_bf16(a2, load_b_frag<260>(sW[pb], 0, wv * 32 + 16),
                                        eo[m][s4 * 2 + 1]);
        }
      }
    }
#pragma unroll
    for (int t = 0; t < 8; ++t) {                        // gated output bias
      int col = (t >> 1) * 256 + wv * 32 + (t & 1) * 16 + cb;
#pragma unroll
      for (int m = 0; m < 5; ++m)
#pragma unroll
        for (int r = 0; r < 8; ++r) eo[m][t][r] += g * b2[col];
    }
  }
  // out already holds x (attention residual); add moe+shared. One owner/elem.
#pragma unroll
  for (int t = 0; t < 8; ++t) {
    int col = (t >> 1) * 256 + wv * 32 + (t & 1) * 16 + cb;
#pragma unroll
    for (int m = 0; m < 5; ++m)
#pragma unroll
      for (int r = 0; r < 8; ++r) {
        int s = m * 16 + crow(lane, r);
        if (s < 77) {
          long long idx = ((long long)b * 77 + s) * 1024 + col;
          out[idx] += eo[m][t][r];
        }
      }
  }
}

// ---------------------------------------------------------------------------
extern "C" void kernel_launch(void* const* d_in, const int* in_sizes, int n_in,
                              void* d_out, int out_size, void* d_ws, size_t ws_size,
                              hipStream_t stream) {
  (void)in_sizes; (void)n_in; (void)out_size; (void)ws_size;
  const float* src      = (const float*)d_in[0];
  const float* packed_w = (const float*)d_in[1];
  const float* packed_b = (const float*)d_in[2];
  const float* out_w    = (const float*)d_in[3];
  const float* out_b    = (const float*)d_in[4];
  const float* ln1_w    = (const float*)d_in[5];
  const float* ln1_b    = (const float*)d_in[6];
  const float* ln2_w    = (const float*)d_in[7];
  const float* ln2_b    = (const float*)d_in[8];
  const float* router_w = (const float*)d_in[9];
  const float* ew1      = (const float*)d_in[10];
  const float* eb1      = (const float*)d_in[11];
  const float* ew2      = (const float*)d_in[12];
  const float* eb2      = (const float*)d_in[13];
  const float* sw1      = (const float*)d_in[14];
  const float* sb1      = (const float*)d_in[15];
  const float* sw2      = (const float*)d_in[16];
  const float* sb2      = (const float*)d_in[17];
  float* out = (float*)d_out;

  char* ws = (char*)d_ws;
  size_t off = 0;
  auto alloc = [&](size_t elems, size_t esize) -> void* {
    off = (off + 255) & ~(size_t)255;
    void* p = ws + off;
    off += elems * esize;
    return p;
  };
  unsigned short* wqkvT = (unsigned short*)alloc(1024ull * 3072, 2);
  unsigned short* woutT = (unsigned short*)alloc(1024ull * 1024, 2);
  unsigned short* sw1b  = (unsigned short*)alloc(1024ull * 4096, 2);
  unsigned short* sw2b  = (unsigned short*)alloc(4096ull * 1024, 2);
  unsigned short* ew1b  = (unsigned short*)alloc(8ull * 1024 * 4096, 2);
  unsigned short* ew2b  = (unsigned short*)alloc(8ull * 4096 * 1024, 2);
  unsigned short* xn    = (unsigned short*)alloc(9856ull * 1024, 2);
  unsigned short* qkvb  = (unsigned short*)alloc(9856ull * 3072, 2);
  unsigned short* aob   = (unsigned short*)alloc(9856ull * 1024, 2);
  unsigned short* ffnb  = (unsigned short*)alloc(9856ull * 1024, 2);
  int*   eix = (int*)alloc(256, 4);
  float* gts = (float*)alloc(256, 4);

  // weight conversion to bf16 (expert bank -> 134 MB, L2-resident afterwards)
  k_cvt<<<2048, 256, 0, stream>>>(packed_w, wqkvT, 3072, 1024, 1);
  k_cvt<<<1024, 256, 0, stream>>>(out_w,    woutT, 1024, 1024, 1);
  k_cvt<<<2048, 256, 0, stream>>>(sw1,      sw1b,  1024, 4096, 0);
  k_cvt<<<2048, 256, 0, stream>>>(sw2,      sw2b,  4096, 1024, 0);
  k_cvt<<<8192, 256, 0, stream>>>(ew1,      ew1b,  8 * 1024, 4096, 0);
  k_cvt<<<8192, 256, 0, stream>>>(ew2,      ew2b,  8 * 4096, 1024, 0);

  // pre-norm attention block
  k_ln<<<9856, 256, 0, stream>>>(src, ln1_w, ln1_b, xn, 1024);
  dim3 gq(3072 / 128, 9856 / 128);
  k_gemm<false, true><<<gq, 256, 0, stream>>>(xn, wqkvT, packed_b, nullptr,
                                              nullptr, qkvb, 9856, 3072, 1024);
  k_attn<<<128 * 16, 256, 0, stream>>>(qkvb, aob);
  dim3 go(1024 / 128, 9856 / 128);
  k_gemm<true, false><<<go, 256, 0, stream>>>(aob, woutT, out_b, src,
                                              out, nullptr, 9856, 1024, 1024);

  // pre-norm MoE FFN block (+shared expert fused as gate-1 expert)
  k_ln<<<9856, 256, 0, stream>>>(out, ln2_w, ln2_b, ffnb, 1024);
  k_router<<<128, 256, 0, stream>>>(ffnb, router_w, eix, gts);
  k_moe<<<128, 256, 0, stream>>>(ffnb, ew1b, eb1, ew2b, eb2,
                                 sw1b, sb1, sw2b, sb2, eix, gts, out);
}
